// SSMaxEmotionSelfAttention_7215545057730
// MI455X (gfx1250) — matrix operand order repaired and checked by reading, not compiled, not run on hardware
//
#include <hip/hip_runtime.h>
#include <hip/hip_bf16.h>

// ---------------------------------------------------------------------------
// SSMax Emotion Self-Attention for MI455X (gfx1250), wave32 + WMMA bf16.
// B=2, T=2048, E=1024, H=16, D=64.
// Pipeline: cvt->bf16 | emo-stats | QKV WMMA GEMMs |
//           fused scores+softmax+PV (LDS-resident row block, TDM-staged V)
//           | out-proj GEMM
// ---------------------------------------------------------------------------

#define B_DIM 2
#define T_DIM 2048
#define E_DIM 1024
#define H_DIM 16
#define D_DIM 64
#define OUT_ELEMS ((size_t)B_DIM * T_DIM * E_DIM)           // 4,194,304
#define QK_SCALE 0.57184642f   // 0.6 * ln(2048) / sqrt(64)
#define EMO_GAMMA_C 0.5f
#define SROW 2052              // LDS row stride (floats): 2048 + 4 pad (bank-conflict-free)
#define SMEM_FLOATS (16 * SROW + 256 + 16 + 8 * 256)
#define VCHUNK 32              // V rows per TDM chunk (4 KB contiguous)

#if defined(__has_builtin)
#if __has_builtin(__builtin_amdgcn_tensor_load_to_lds) && __has_builtin(__builtin_amdgcn_s_wait_tensorcnt)
#define HAVE_TDM 1
#endif
#endif
#ifndef HAVE_TDM
#define HAVE_TDM 0
#endif

typedef __attribute__((ext_vector_type(16))) __bf16          v16bf;
typedef __attribute__((ext_vector_type(16))) unsigned short  v16u;
typedef __attribute__((ext_vector_type(8)))  float           v8f;
typedef __attribute__((ext_vector_type(4)))  unsigned int    u32x4;
typedef __attribute__((ext_vector_type(8)))  int             i32x8;
typedef __attribute__((ext_vector_type(4)))  int             i32x4;

__device__ __forceinline__ unsigned short f32_to_bf16(float f) {
    unsigned int u = __float_as_uint(f);
    unsigned int r = u + 0x7FFFu + ((u >> 16) & 1u);   // round to nearest even
    return (unsigned short)(r >> 16);
}

#if HAVE_TDM
// 1-D contiguous TDM copy: nelem bf16 elements global -> LDS (D# per ISA §8).
__device__ __forceinline__ void tdm_load_1d(unsigned lds_off, const void* gptr, unsigned nelem) {
    unsigned long long ga = (unsigned long long)(size_t)gptr;
    u32x4 g0;
    g0[0] = 1u;                                             // count=1 (valid), user mode
    g0[1] = lds_off;                                        // lds_addr (bytes)
    g0[2] = (unsigned)(ga & 0xffffffffu);                   // global_addr[31:0]
    g0[3] = (unsigned)((ga >> 32) & 0x1ffffffu) | (2u << 30); // addr[56:32] | type=2
    i32x8 g1;
    g1[0] = (1 << 16);                                      // wg_mask=0, data_size=1 (2B)
    g1[1] = (int)((nelem & 0xffffu) << 16);                 // tensor_dim0[15:0]
    g1[2] = (int)(((nelem >> 16) & 0xffffu) | (1u << 16));  // tensor_dim0[31:16], tensor_dim1=1
    g1[3] = (int)(nelem << 16);                             // tile_dim0 (<=65535)
    g1[4] = 0;                                              // tile_dim1=0 (1-D), tile_dim2=0
    g1[5] = (int)nelem;                                     // tensor_dim0_stride[31:0]
    g1[6] = 0;
    g1[7] = 0;
    i32x4 z4 = {0, 0, 0, 0};
#if __clang_major__ >= 23
    i32x8 z8 = {0, 0, 0, 0, 0, 0, 0, 0};
    __builtin_amdgcn_tensor_load_to_lds(g0, g1, z4, z4, z8, 0);
#else
    __builtin_amdgcn_tensor_load_to_lds(g0, g1, z4, z4, 0);
#endif
}
#endif

// A-matrix fragment, 16x32 bf16, row-major source, row stride ld (elements).
// Layout: lane m = L&15, half = L>>4; VGPR v: K=2v+16*half;
// pairs packed per VGPR.
__device__ __forceinline__ v16bf load_a_bf16(const unsigned short* __restrict__ src, int ld) {
    int lane = threadIdx.x & 31;
    int m = lane & 15, half = lane >> 4;
    const unsigned short* p = src + (size_t)m * ld + 16 * half;
    v16u u;
#pragma unroll
    for (int v = 0; v < 8; ++v) {
        int k = 2 * v;
        unsigned int w = *(const unsigned int*)(p + k);      // dword load: 2 bf16
        u[2 * v]     = (unsigned short)(w & 0xffffu);
        u[2 * v + 1] = (unsigned short)(w >> 16);
    }
    return __builtin_bit_cast(v16bf, u);
}

// Same A fragment from an fp32 LDS tile (convert on the fly) — P tiles.
__device__ __forceinline__ v16bf load_a_lds_f32(const float* p0, int ld) {
    int lane = threadIdx.x & 31;
    int m = lane & 15, half = lane >> 4;
    const float* p = p0 + m * ld + 16 * half;
    v16u u;
#pragma unroll
    for (int v = 0; v < 8; ++v) {
        int k = 2 * v;
        u[2 * v]     = f32_to_bf16(p[k]);
        u[2 * v + 1] = f32_to_bf16(p[k + 1]);
    }
    return __builtin_bit_cast(v16bf, u);
}

// B-matrix fragment 32x16 where B_wmma[k][n] = src[n*ld + k]  (src = rows of W
// or K-matrix, row-major over k). Lane n = L&15, half = L>>4; K = 2v+16*half.
__device__ __forceinline__ v16bf load_bt_bf16(const unsigned short* __restrict__ src, int ld) {
    int lane = threadIdx.x & 31;
    int n = lane & 15, half = lane >> 4;
    const unsigned short* p = src + (size_t)n * ld + 16 * half;
    v16u u;
#pragma unroll
    for (int v = 0; v < 8; ++v) {
        unsigned int w = *(const unsigned int*)(p + 2 * v);  // dword load
        u[2 * v]     = (unsigned short)(w & 0xffffu);
        u[2 * v + 1] = (unsigned short)(w >> 16);
    }
    return __builtin_bit_cast(v16bf, u);
}

// B-matrix fragment 32x16 from a true row-major KxN source (V rows).
__device__ __forceinline__ v16bf load_b_rm_bf16(const unsigned short* src, int ld) {
    int lane = threadIdx.x & 31;
    int n = lane & 15, half = lane >> 4;
    const unsigned short* p = src + n + (size_t)(16 * half) * ld;
    v16u u;
#pragma unroll
    for (int v = 0; v < 8; ++v) {
        u[2 * v]     = p[(size_t)(2 * v) * ld];
        u[2 * v + 1] = p[(size_t)(2 * v + 1) * ld];
    }
    return __builtin_bit_cast(v16bf, u);
}

__device__ __forceinline__ v8f wmma_bf16(v16bf a, v16bf b, v8f c) {
    return __builtin_amdgcn_wmma_f32_16x16x32_bf16(false, a, false, b, (short)0, c, false, false);
}

// ---------------------------------------------------------------------------
// fp32 -> bf16 conversion
// ---------------------------------------------------------------------------
__global__ __launch_bounds__(256) void cvt_bf16_kernel(const float* __restrict__ in,
                                                       unsigned short* __restrict__ out, int n) {
    int i = blockIdx.x * 256 + threadIdx.x;
    if (i < n) out[i] = f32_to_bf16(in[i]);
}

// ---------------------------------------------------------------------------
// Emotion bias stats, closed form (geometric sums of the exp kernel):
//   stats[2b] = gamma/std, stats[2b+1] = -mean*gamma/std  (ddof=1)
// ---------------------------------------------------------------------------
__global__ __launch_bounds__(256) void emo_stats_kernel(const float* __restrict__ emotion,
                                                        float* __restrict__ stats) {
    const float C1 = 7.510413f;   // a/(1-a),   a = exp(-1/8)
    const float C2 = 3.520816f;   // a2/(1-a2), a2 = exp(-1/4)
    int b = blockIdx.x;
    int t = threadIdx.x;
    float sum1 = 0.f, sum2 = 0.f;
    for (int i = t; i < T_DIM; i += 256) {
        float e = emotion[b * T_DIM + i];
        float nl = (float)i, nr = (float)(T_DIM - 1 - i);
        float r = 1.f + C1 * (1.f - __expf(-0.125f * nl)) + C1 * (1.f - __expf(-0.125f * nr));
        float s = 1.f + C2 * (1.f - __expf(-0.25f  * nl)) + C2 * (1.f - __expf(-0.25f  * nr));
        sum1 += e * r;
        sum2 += e * e * s;
    }
    __shared__ float ra[256], rb[256];
    ra[t] = sum1; rb[t] = sum2;
    __syncthreads();
    for (int s = 128; s > 0; s >>= 1) {
        if (t < s) { ra[t] += ra[t + s]; rb[t] += rb[t + s]; }
        __syncthreads();
    }
    if (t == 0) {
        float n = (float)T_DIM * (float)T_DIM;
        float mean = ra[0] / n;
        float var = (rb[0] - n * mean * mean) / (n - 1.f);
        float sd = sqrtf(fmaxf(var, 0.f));
        sd = fmaxf(sd, 1e-6f);
        float scale = EMO_GAMMA_C / sd;
        stats[2 * b]     = scale;
        stats[2 * b + 1] = -mean * scale;
    }
}

// ---------------------------------------------------------------------------
// GEMM: Y[m,n] = sum_k A[m,k] * W[n,k] + bias[n]   (A,W bf16; acc fp32)
// Wave tile 32x64, workgroup (8 waves) tile 64x256.
// mode 0: fp32 row-major [M,N] to outF ; mode 1: bf16 [B,H,T,D] scatter.
// ---------------------------------------------------------------------------
__global__ __launch_bounds__(256) void gemm_bf16_wmma(const unsigned short* __restrict__ A,
                                                      const unsigned short* __restrict__ W,
                                                      const float* __restrict__ bias,
                                                      int K, int N,
                                                      float* __restrict__ outF,
                                                      unsigned short* __restrict__ outB,
                                                      int mode) {
    int wave  = threadIdx.x >> 5;
    int waveM = wave & 1, waveN = wave >> 1;
    int m0 = blockIdx.x * 64 + waveM * 32;
    int n0 = blockIdx.y * 256 + waveN * 64;

    v8f acc[2][4];
#pragma unroll
    for (int mi = 0; mi < 2; ++mi)
#pragma unroll
        for (int ni = 0; ni < 4; ++ni) acc[mi][ni] = (v8f){};

    for (int k0 = 0; k0 < K; k0 += 32) {
        if (k0 + 32 < K) {   // speculative prefetch of next K tile
            __builtin_prefetch(A + (size_t)m0 * K + k0 + 32, 0, 3);
            __builtin_prefetch(W + (size_t)n0 * K + k0 + 32, 0, 3);
        }
        v16bf a0 = load_a_bf16(A + (size_t)m0 * K + k0, K);
        v16bf a1 = load_a_bf16(A + (size_t)(m0 + 16) * K + k0, K);
        v16bf bfr[4];
#pragma unroll
        for (int ni = 0; ni < 4; ++ni)
            bfr[ni] = load_bt_bf16(W + (size_t)(n0 + 16 * ni) * K + k0, K);
#pragma unroll
        for (int ni = 0; ni < 4; ++ni) {
            acc[0][ni] = wmma_bf16(a0, bfr[ni], acc[0][ni]);
            acc[1][ni] = wmma_bf16(a1, bfr[ni], acc[1][ni]);
        }
    }

    int lane = threadIdx.x & 31;
    int jj = lane & 15, half = lane >> 4;
#pragma unroll
    for (int mi = 0; mi < 2; ++mi) {
#pragma unroll
        for (int ni = 0; ni < 4; ++ni) {
#pragma unroll
            for (int v = 0; v < 8; ++v) {
                int i = m0 + 16 * mi + v + 8 * half;
                int j = n0 + 16 * ni + jj;
                float val = acc[mi][ni][v] + bias[j];
                if (mode == 0) {
                    outF[(size_t)i * N + j] = val;
                } else {
                    int bb = i >> 11, t = i & (T_DIM - 1);
                    int h  = j >> 6,  d = j & (D_DIM - 1);
                    outB[((((size_t)bb * H_DIM + h) * T_DIM) + t) * D_DIM + d] = f32_to_bf16(val);
                }
            }
        }
    }
}

// ---------------------------------------------------------------------------
// Fused attention: one workgroup = one (b,h) x 16-row query block.
//   phase 1: biased scores (WMMA) -> LDS (16 x SROW fp32)
//   phase 2: row softmax in LDS; single coalesced global write of attn
//   phase 3: PV from LDS probs; V rows staged into LDS by the Tensor Data
//            Mover (1-D D#, double-buffered per K-half, TENSORcnt-tracked),
//            cross-wave partial reduction, bf16 ctx out.
// grid = (T/16, 1, B*H), 256 threads (8 waves).
// Dyn LDS: SMEM_FLOATS fp32 + 4x(32x64) bf16 V buffers (~157 KB -> 2 WGs/WGP)
// ---------------------------------------------------------------------------
__global__ __launch_bounds__(256) void attn_fused_kernel(const unsigned short* __restrict__ Q,
                                                         const unsigned short* __restrict__ Km,
                                                         const unsigned short* __restrict__ V,
                                                         const float* __restrict__ emotion,
                                                         const float* __restrict__ stats,
                                                         float* __restrict__ attn,
                                                         unsigned short* __restrict__ ctx) {
    extern __shared__ float smem[];
    float* Sld  = smem;                 // 16 x SROW  scores -> probs
    float* red  = Sld + 16 * SROW;      // 16 x 16 reduction tile
    float* sm   = red + 256;            // 16 per-row 1/sum
    float* Pred = sm + 16;              // 8 x (16x16) partial ctx
    unsigned short* Vld = (unsigned short*)(Pred + 2048);   // 4 x VCHUNK*64 bf16

    const int z = blockIdx.z;           // b*H + h
    const int b = z >> 4, h = z & 15;
    const int i0 = blockIdx.x * 16;
    const int t = threadIdx.x;
    const int wave = t >> 5;
    const int lane = t & 31;
    const int jj = lane & 15, half = lane >> 4;

    const unsigned short* Qb = Q  + (size_t)z * T_DIM * D_DIM;
    const unsigned short* Kb = Km + (size_t)z * T_DIM * D_DIM;
    const unsigned short* Vb = V  + (size_t)z * T_DIM * D_DIM;

    // ---- phase 1: scores + bias -> LDS ------------------------------------
    v16bf aq0 = load_a_bf16(Qb + (size_t)i0 * D_DIM + 0,  D_DIM);
    v16bf aq1 = load_a_bf16(Qb + (size_t)i0 * D_DIM + 32, D_DIM);

    const float scale = stats[2 * b], shift = stats[2 * b + 1];
    float ev[8]; int ivg[8];
#pragma unroll
    for (int v = 0; v < 8; ++v) {
        ivg[v] = i0 + v + 8 * half;
        ev[v] = emotion[b * T_DIM + ivg[v]];
    }

    for (int c = 0; c < 4; ++c) {
        int jbase = wave * 256 + c * 64;
        v8f acc[4];
#pragma unroll
        for (int ni = 0; ni < 4; ++ni) acc[ni] = (v8f){};
#pragma unroll
        for (int ni = 0; ni < 4; ++ni) {
            const unsigned short* krow = Kb + (size_t)(jbase + 16 * ni) * D_DIM;
            acc[ni] = wmma_bf16(aq0, load_bt_bf16(krow + 0,  D_DIM), acc[ni]);
            acc[ni] = wmma_bf16(aq1, load_bt_bf16(krow + 32, D_DIM), acc[ni]);
        }
#pragma unroll
        for (int ni = 0; ni < 4; ++ni) {
            int j = jbase + 16 * ni + jj;
#pragma unroll
            for (int v = 0; v < 8; ++v) {
                float dist = fabsf((float)(ivg[v] - j));
                float s = acc[ni][v] * QK_SCALE + ev[v] * __expf(-0.125f * dist) * scale + shift;
                Sld[(v + 8 * half) * SROW + j] = s;
            }
        }
    }
    __syncthreads();

    // ---- phase 2: row softmax in LDS, one coalesced attn write ------------
    const int r = t >> 4, g = t & 15;   // 16 threads cooperate per row
    float lmax = -1e30f;
#pragma unroll 8
    for (int k = 0; k < 128; ++k) lmax = fmaxf(lmax, Sld[r * SROW + g + 16 * k]);
    red[t] = lmax;
    __syncthreads();
    float rowmax = -1e30f;
#pragma unroll
    for (int gg = 0; gg < 16; ++gg) rowmax = fmaxf(rowmax, red[r * 16 + gg]);

    float lsum = 0.f;
#pragma unroll 8
    for (int k = 0; k < 128; ++k) {
        int idx = r * SROW + g + 16 * k;
        float e = __expf(Sld[idx] - rowmax);
        Sld[idx] = e;
        lsum += e;
    }
    __syncthreads();                    // red reads (rowmax) done before overwrite
    red[t] = lsum;
    __syncthreads();
    float rowsum = 0.f;
#pragma unroll
    for (int gg = 0; gg < 16; ++gg) rowsum += red[r * 16 + gg];
    if (g == 0) sm[r] = 1.f / rowsum;
    __syncthreads();

    float* attnBlk = attn + ((size_t)z * T_DIM + i0) * T_DIM;
#pragma unroll 8
    for (int q = 0; q < 128; ++q) {     // linear sweep: coalesced global write
        int idx = t + 256 * q;
        int row = idx >> 11, col = idx & (T_DIM - 1);
        float val = Sld[row * SROW + col] * sm[row];
        Sld[row * SROW + col] = val;
        attnBlk[(size_t)row * T_DIM + col] = val;
    }
    __syncthreads();

    // ---- phase 3: ctx = P @ V from LDS probs ------------------------------
    const int dstrip = (wave & 3) * 16;         // 4 d-strips of 16
    const int halfK  = wave >> 2;               // 2 K-halves
    const int jstart = halfK * 1024;
    v8f cacc = (v8f){};

#if HAVE_TDM
    // Double-buffered TDM staging of contiguous V row chunks (per K-half).
    const int NCH = 1024 / VCHUNK;              // 32 chunks of 32 rows
    if ((wave & 3) == 0) {                      // waves 0 and 4 issue for their half
        unsigned off0 = (unsigned)(size_t)(void*)(Vld + (size_t)(halfK * 2) * (VCHUNK * D_DIM));
        tdm_load_1d(off0, Vb + (size_t)jstart * D_DIM, VCHUNK * D_DIM);
        __builtin_amdgcn_s_wait_tensorcnt(0);
    }
    __syncthreads();
    for (int cchunk = 0; cchunk < NCH; ++cchunk) {
        int j0 = jstart + cchunk * VCHUNK;
        int par = cchunk & 1;
        if (((wave & 3) == 0) && (cchunk + 1 < NCH)) {
            unsigned offn = (unsigned)(size_t)(void*)
                (Vld + (size_t)(halfK * 2 + (1 - par)) * (VCHUNK * D_DIM));
            tdm_load_1d(offn, Vb + (size_t)(j0 + VCHUNK) * D_DIM, VCHUNK * D_DIM);
        }
        const unsigned short* vb = Vld + (size_t)(halfK * 2 + par) * (VCHUNK * D_DIM);
        v16bf a   = load_a_lds_f32(Sld + j0, SROW);
        v16bf bfr = load_b_rm_bf16(vb + dstrip, D_DIM);
        cacc = wmma_bf16(a, bfr, cacc);
        if (((wave & 3) == 0) && (cchunk + 1 < NCH)) __builtin_amdgcn_s_wait_tensorcnt(0);
        __syncthreads();
    }
#else
    for (int j0 = jstart; j0 < jstart + 1024; j0 += 32) {
        v16bf a   = load_a_lds_f32(Sld + j0, SROW);
        v16bf bfr = load_b_rm_bf16(Vb + (size_t)j0 * D_DIM + dstrip, D_DIM);
        cacc = wmma_bf16(a, bfr, cacc);
    }
#endif

#pragma unroll
    for (int v = 0; v < 8; ++v)
        Pred[wave * 256 + (v + 8 * half) * 16 + jj] = cacc[v];
    __syncthreads();

#pragma unroll
    for (int q = 0; q < 4; ++q) {       // reduce K-halves, write bf16 ctx
        int e = t + 256 * q;
        int i = e >> 6, d = e & 63;
        int strip = d >> 4, dj = d & 15;
        float val = Pred[strip * 256 + i * 16 + dj] + Pred[(strip + 4) * 256 + i * 16 + dj];
        ctx[((size_t)b * T_DIM + i0 + i) * E_DIM + h * D_DIM + d] = f32_to_bf16(val);
    }
}

// ---------------------------------------------------------------------------
// Host launch
// ---------------------------------------------------------------------------
extern "C" void kernel_launch(void* const* d_in, const int* in_sizes, int n_in,
                              void* d_out, int out_size, void* d_ws, size_t ws_size,
                              hipStream_t stream) {
    (void)in_sizes; (void)n_in; (void)out_size; (void)ws_size;

    const float* x       = (const float*)d_in[0];
    const float* emotion = (const float*)d_in[1];
    const float* Wq = (const float*)d_in[2];
    const float* bq = (const float*)d_in[3];
    const float* Wk = (const float*)d_in[4];
    const float* bk = (const float*)d_in[5];
    const float* Wv = (const float*)d_in[6];
    const float* bv = (const float*)d_in[7];
    const float* Wo = (const float*)d_in[8];
    const float* bo = (const float*)d_in[9];

    float* outF = (float*)d_out;
    float* attn = outF + OUT_ELEMS;

    const size_t MB = 1024 * 1024;
    char* ws = (char*)d_ws;
    unsigned short* xb   = (unsigned short*)(ws);             // 8 MB (reused as ctx later)
    unsigned short* Wqb  = (unsigned short*)(ws + 8  * MB);
    unsigned short* Wkb  = (unsigned short*)(ws + 10 * MB);
    unsigned short* Wvb  = (unsigned short*)(ws + 12 * MB);
    unsigned short* Wob  = (unsigned short*)(ws + 14 * MB);
    unsigned short* Qb   = (unsigned short*)(ws + 16 * MB);   // [B,H,T,D] bf16
    unsigned short* Kb   = (unsigned short*)(ws + 24 * MB);
    unsigned short* Vb   = (unsigned short*)(ws + 32 * MB);
    float*          stat = (float*)(ws + 40 * MB);
    unsigned short* ctxb = xb;                                // x dead after QKV GEMMs

    const int M  = B_DIM * T_DIM;        // 4096
    const int nX = M * E_DIM;
    const int nW = E_DIM * E_DIM;

    // 1) fp32 -> bf16 conversions
    cvt_bf16_kernel<<<(nX + 255) / 256, 256, 0, stream>>>(x,  xb,  nX);
    cvt_bf16_kernel<<<(nW + 255) / 256, 256, 0, stream>>>(Wq, Wqb, nW);
    cvt_bf16_kernel<<<(nW + 255) / 256, 256, 0, stream>>>(Wk, Wkb, nW);
    cvt_bf16_kernel<<<(nW + 255) / 256, 256, 0, stream>>>(Wv, Wvb, nW);
    cvt_bf16_kernel<<<(nW + 255) / 256, 256, 0, stream>>>(Wo, Wob, nW);

    // 2) emotion bias stats
    emo_stats_kernel<<<B_DIM, 256, 0, stream>>>(emotion, stat);

    // 3) Q/K/V projections -> bf16 [B,H,T,D]
    dim3 gg(M / 64, E_DIM / 256);
    gemm_bf16_wmma<<<gg, 256, 0, stream>>>(xb, Wqb, bq, E_DIM, E_DIM, nullptr, Qb, 1);
    gemm_bf16_wmma<<<gg, 256, 0, stream>>>(xb, Wkb, bk, E_DIM, E_DIM, nullptr, Kb, 1);
    gemm_bf16_wmma<<<gg, 256, 0, stream>>>(xb, Wvb, bv, E_DIM, E_DIM, nullptr, Vb, 1);

    // 4) fused scores + softmax + PV (LDS-resident 16-row blocks, TDM-staged V)
    const size_t smem_bytes = (size_t)SMEM_FLOATS * sizeof(float)
                            + (size_t)4 * VCHUNK * D_DIM * sizeof(unsigned short);
    hipFuncSetAttribute((const void*)attn_fused_kernel,
                        hipFuncAttributeMaxDynamicSharedMemorySize, (int)smem_bytes);
    attn_fused_kernel<<<dim3(T_DIM / 16, 1, B_DIM * H_DIM), 256, smem_bytes, stream>>>(
        Qb, Kb, Vb, emotion, stat, attn, ctxb);

    // 5) out = ctx @ Wo^T + bo -> fp32 d_out
    gemm_bf16_wmma<<<gg, 256, 0, stream>>>(ctxb, Wob, bo, E_DIM, E_DIM, outF, nullptr, 0);
}
